// _PAM_4629974745542
// MI455X (gfx1250) — compile-verified, matrix operand order checked by reading, not run on hardware
//
#include <hip/hip_runtime.h>

// ---------- CDNA5 (gfx1250) wave32 WMMA types ----------
typedef _Float16 f16;
typedef __attribute__((ext_vector_type(16))) _Float16 v16h;
typedef __attribute__((ext_vector_type(8)))  _Float16 v8h;
typedef __attribute__((ext_vector_type(8)))  float    v8f;
typedef __attribute__((ext_vector_type(4)))  float    v4f;

#define K_BATCH 4
#define K_C     512
#define K_CO    64      /* C/8 */
#define K_N     4096    /* H*W */

static __device__ __forceinline__ int lane_id() { return (int)(threadIdx.x & 31u); }

static __device__ __forceinline__ v8f wmma_f16(v16h a, v16h b, v8f c) {
  // v_wmma_f32_16x16x32_f16 : D = A(16x32) * B(32x16) + C(16x16 f32)
  return __builtin_amdgcn_wmma_f32_16x16x32_f16(false, a, false, b, (short)0, c, false, false);
}

// ---- DPP 16-lane row reductions (no LDS, no dscnt stalls) ----
// ctrl: 0xB1 = quad_perm(1,0,3,2)=xor1, 0x4E = quad_perm(2,3,0,1)=xor2,
//       0x124 = row_ror:4, 0x128 = row_ror:8 (rows are 16 lanes in wave32)
template <int CTRL>
static __device__ __forceinline__ float dpp_mov(float v) {
  return __int_as_float(__builtin_amdgcn_update_dpp(
      __float_as_int(v), __float_as_int(v), CTRL, 0xf, 0xf, true));
}
static __device__ __forceinline__ float row16_max(float v) {
  v = fmaxf(v, dpp_mov<0x0B1>(v));
  v = fmaxf(v, dpp_mov<0x04E>(v));
  v = fmaxf(v, dpp_mov<0x124>(v));
  v = fmaxf(v, dpp_mov<0x128>(v));
  return v;
}
static __device__ __forceinline__ float row16_sum(float v) {
  v += dpp_mov<0x0B1>(v);
  v += dpp_mov<0x04E>(v);
  v += dpp_mov<0x124>(v);
  v += dpp_mov<0x128>(v);
  return v;
}

// A-matrix 16x32 f16 wave32 layout (ISA 7.12.2):
// lanes 0-15 : M=lane,    K = {0..7} then {16..23}
// lanes 16-31: M=lane-16, K = {8..15} then {24..31}
static __device__ __forceinline__ v16h load_a(const f16* __restrict__ p, int ms, int ks) {
  const int l  = lane_id();
  const int m  = l & 15;
  const int kb = (l >> 4) << 3;
  v16h a;
#pragma unroll
  for (int j = 0; j < 4; ++j) {
    const int k0 = kb + 2 * j;
    a[2*j]       = p[m*ms + (k0    )*ks];
    a[2*j+1]     = p[m*ms + (k0 + 1)*ks];
    a[8 + 2*j]   = p[m*ms + (16 + k0)*ks];
    a[8 + 2*j+1] = p[m*ms + (17 + k0)*ks];
  }
  return a;
}

// B-matrix 32x16 f16 wave32 layout:
// lanes 0-15 : N=lane,    K = {0..7} then {16..23}
// lanes 16-31: N=lane-16, K = {8..15} then {24..31}
static __device__ __forceinline__ v16h load_b(const f16* __restrict__ p, int ks, int ns) {
  const int l  = lane_id();
  const int n  = l & 15;
  const int kb = (l >> 4) << 3;
  v16h b;
#pragma unroll
  for (int j = 0; j < 4; ++j) {
    const int k0 = kb + 2 * j;
    b[2*j]       = p[(k0    )*ks + n*ns];
    b[2*j+1]     = p[(k0 + 1)*ks + n*ns];
    b[8 + 2*j]   = p[(16 + k0)*ks + n*ns];
    b[8 + 2*j+1] = p[(17 + k0)*ks + n*ns];
  }
  return b;
}

// Same as load_b but source is fp32 (weights), converted on the fly.
static __device__ __forceinline__ v16h load_b_f32(const float* __restrict__ p, int ks, int ns) {
  const int l  = lane_id();
  const int n  = l & 15;
  const int kb = (l >> 4) << 3;
  v16h b;
#pragma unroll
  for (int j = 0; j < 4; ++j) {
    const int k0 = kb + 2 * j;
    b[2*j]       = (f16)p[(k0    )*ks + n*ns];
    b[2*j+1]     = (f16)p[(k0 + 1)*ks + n*ns];
    b[8 + 2*j]   = (f16)p[(16 + k0)*ks + n*ns];
    b[8 + 2*j+1] = (f16)p[(17 + k0)*ks + n*ns];
  }
  return b;
}

// ---------------------------------------------------------------------------
// Projection: feat[b, n, o] = sum_c W[o,c] * x[b,c,n] + bias[o]   (stored f16)
// TRANS=false: out[b][n][o]  (Q/K layout: head-dim contiguous for A/B frags)
// TRANS=true : out[b][o][n]  (V layout: position contiguous for B frags)
// Grid: (N/128, O/64, B). Block: 256 (8 waves). Wave w owns n-strip w*16..+15.
// ---------------------------------------------------------------------------
template <bool TRANS>
__global__ __launch_bounds__(256) void proj_f16_kernel(
    const float* __restrict__ x, const float* __restrict__ W,
    const float* __restrict__ bias, f16* __restrict__ out, int O) {
  const int b    = blockIdx.z;
  const int n0   = blockIdx.x * 128;
  const int o0   = blockIdx.y * 64;
  const int tid  = (int)threadIdx.x;
  const int wave = tid >> 5;
  const int l    = tid & 31;

  __shared__ f16 As[128][34];   // x^T tile [n][k], padded

  const float* xb = x + (size_t)b * K_C * K_N;
  v8f acc[4] = {};

  for (int cb = 0; cb < K_C; cb += 32) {
    // Stage x[cb..cb+31][n0..n0+127] -> LDS transposed as f16.
    {
      const int kk = tid >> 3;          // 0..31 (channel within block)
      const int nn = (tid & 7) * 16;    // column chunk
      const float* src = xb + (size_t)(cb + kk) * K_N + n0 + nn;
#pragma unroll
      for (int t = 0; t < 16; ++t) As[nn + t][kk] = (f16)src[t];
    }
    __syncthreads();

    const v16h a = load_a(&As[wave * 16][0], 34, 1);
#pragma unroll
    for (int s = 0; s < 4; ++s) {
      // B(k,o) = W[(o0+s*16+o)*C + cb + k]  -> ks=1 (contiguous), ns=C
      const v16h bf = load_b_f32(W + (size_t)(o0 + s * 16) * K_C + cb, 1, K_C);
      acc[s] = wmma_f16(a, bf, acc[s]);
    }
    __syncthreads();
  }

  // Epilogue: bias add, f16 store.
  const int rb = (l >> 4) << 3;
  const int nc = l & 15;
  if (TRANS) {
    // out[b][o][n]: lane's 8 rows are 8 consecutive n -> one v8h store per tile
    f16* ob = out + (size_t)b * (size_t)O * K_N;
    const int qbase = n0 + wave * 16 + rb;   // multiple of 8 -> 16B aligned
#pragma unroll
    for (int s = 0; s < 4; ++s) {
      const int o = o0 + s * 16 + nc;
      const float bv = bias[o];
      v8h pk;
#pragma unroll
      for (int j = 0; j < 8; ++j) pk[j] = (f16)(acc[s][j] + bv);
      *(v8h*)(ob + (size_t)o * K_N + qbase) = pk;
    }
  } else {
    f16* ob = out + (size_t)b * K_N * O;
#pragma unroll
    for (int s = 0; s < 4; ++s) {
      const float bv = bias[o0 + s * 16 + nc];
#pragma unroll
      for (int j = 0; j < 8; ++j) {
        const int q = n0 + wave * 16 + rb + j;
        ob[(size_t)q * O + o0 + s * 16 + nc] = (f16)(acc[s][j] + bv);
      }
    }
  }
}

// ---------------------------------------------------------------------------
// Fused flash-style attention:
//   S[q,m] = sum_k Q[q,k]*K[m,k]; P = softmax_m(S);  E[q,c] = sum_m P[q,m]*V[m,c]
//   out[b,c,q] = alpha * E[q,c] + x[b,c,q]
// Layouts: Q,K = [B][N][64] (k contiguous); V = [B][C][N] (m contiguous).
// Grid: (N/32, B). Block 256 = 8 waves: wave = qw*4 + cw;
//   qw in {0,1}: 16-query strip; cw in {0..3}: 128-wide c slice.
// ---------------------------------------------------------------------------
__global__ __launch_bounds__(256) void pam_attn_kernel(
    const f16* __restrict__ Q, const f16* __restrict__ Kf, const f16* __restrict__ V,
    const float* __restrict__ x, const float* __restrict__ alpha_p,
    float* __restrict__ out) {
  const int b    = blockIdx.y;
  const int tid  = (int)threadIdx.x;
  const int wave = tid >> 5;
  const int qw   = wave >> 2;
  const int cw   = wave & 3;
  const int l    = tid & 31;
  const int rb   = (l >> 4) << 3;
  const int nc   = l & 15;

  const int q0 = blockIdx.x * 32 + qw * 16;
  const int c0 = cw * 128;

  const f16* Qb = Q  + (size_t)b * K_N * K_CO;
  const f16* Kb = Kf + (size_t)b * K_N * K_CO;
  const f16* Vb = V  + (size_t)b * K_C * K_N;   // [C][N]

  __shared__ f16 Pl[8][16][72];   // per-wave P staging (C-layout -> A-layout)

  // Q fragments are loop-invariant: preload both K=32 steps of the head dim.
  v16h aq[2];
  aq[0] = load_a(Qb + (size_t)q0 * K_CO + 0,  K_CO, 1);
  aq[1] = load_a(Qb + (size_t)q0 * K_CO + 32, K_CO, 1);

  float mrow[8], lrow[8];
#pragma unroll
  for (int j = 0; j < 8; ++j) { mrow[j] = -3.0e38f; lrow[j] = 0.0f; }
  v8f accO[8] = {};

  for (int m0 = 0; m0 < K_N; m0 += 64) {
    // ---- S = Q * K^T for 16 q x 64 m (4 WMMA tiles x 2 k-steps) ----
    v8f sacc[4] = {};
#pragma unroll
    for (int t = 0; t < 4; ++t) {
      const f16* kb0 = Kb + (size_t)(m0 + t * 16) * K_CO;
      sacc[t] = wmma_f16(aq[0], load_b(kb0 + 0,  1, K_CO), sacc[t]);
      sacc[t] = wmma_f16(aq[1], load_b(kb0 + 32, 1, K_CO), sacc[t]);
    }

    // ---- online softmax over m (rows span 16 lanes; DPP reductions) ----
    float scale[8];
#pragma unroll
    for (int j = 0; j < 8; ++j) {
      float v = fmaxf(fmaxf(sacc[0][j], sacc[1][j]), fmaxf(sacc[2][j], sacc[3][j]));
      v = row16_max(v);
      const float mn = fmaxf(mrow[j], v);
      scale[j] = __expf(mrow[j] - mn);
      mrow[j]  = mn;
      float ssum = 0.0f;
#pragma unroll
      for (int t = 0; t < 4; ++t) {
        const float p = __expf(sacc[t][j] - mn);
        sacc[t][j] = p;
        ssum += p;
      }
      ssum = row16_sum(ssum);
      lrow[j] = lrow[j] * scale[j] + ssum;
    }

    // rescale running O accumulators
#pragma unroll
    for (int s = 0; s < 8; ++s)
#pragma unroll
      for (int j = 0; j < 8; ++j) accO[s][j] *= scale[j];

    // ---- P: C-layout -> LDS -> A-layout (per-wave private region) ----
#pragma unroll
    for (int t = 0; t < 4; ++t)
#pragma unroll
      for (int j = 0; j < 8; ++j)
        Pl[wave][rb + j][t * 16 + nc] = (f16)sacc[t][j];

    // ---- E += P * V  (16 q x 128 c, 8 tiles x 2 k-steps) ----
    // V is [C][N]: B(k,n) = Vb[(c0+s*16+n)*N + m0+ks*32+k] -> ks=1 (contiguous)
#pragma unroll
    for (int ks = 0; ks < 2; ++ks) {
      const v16h pa = load_a(&Pl[wave][0][ks * 32], 72, 1);
      const f16* vb0 = Vb + (size_t)(c0) * K_N + m0 + ks * 32;
#pragma unroll
      for (int s = 0; s < 8; ++s)
        accO[s] = wmma_f16(pa, load_b(vb0 + (size_t)(s * 16) * K_N, 1, K_N), accO[s]);
    }
  }

  // ---- epilogue: out[b,c,q] = alpha * accO/l + x[b,c,q] ----
  // Streaming phase: x read here is its last use, and `out` is write-once.
  // Use non-temporal (th:NT) b128 accesses so the Q/K/V working set stays
  // resident in L2 for the other attention blocks.
  const float alpha = *alpha_p;
  float inv[8];
#pragma unroll
  for (int j = 0; j < 8; ++j) inv[j] = alpha / lrow[j];
  const float* xb = x + (size_t)b * K_C * K_N;
  float* ob = out + (size_t)b * K_C * K_N;
  const int qb = q0 + rb;   // multiple of 8 -> 32B aligned fp32 addresses
#pragma unroll
  for (int s = 0; s < 8; ++s) {
    const int c = c0 + s * 16 + nc;
    const v4f* px = (const v4f*)(xb + (size_t)c * K_N + qb);
    v4f*       po = (v4f*)(ob + (size_t)c * K_N + qb);
    const v4f x0 = __builtin_nontemporal_load(px);
    const v4f x1 = __builtin_nontemporal_load(px + 1);
    v4f o0, o1;
    o0[0] = accO[s][0] * inv[0] + x0[0];
    o0[1] = accO[s][1] * inv[1] + x0[1];
    o0[2] = accO[s][2] * inv[2] + x0[2];
    o0[3] = accO[s][3] * inv[3] + x0[3];
    o1[0] = accO[s][4] * inv[4] + x1[0];
    o1[1] = accO[s][5] * inv[5] + x1[1];
    o1[2] = accO[s][6] * inv[6] + x1[2];
    o1[3] = accO[s][7] * inv[7] + x1[3];
    __builtin_nontemporal_store(o0, po);
    __builtin_nontemporal_store(o1, po + 1);
  }
}

// ---------------------------------------------------------------------------
extern "C" void kernel_launch(void* const* d_in, const int* in_sizes, int n_in,
                              void* d_out, int out_size, void* d_ws, size_t ws_size,
                              hipStream_t stream) {
  (void)in_sizes; (void)n_in; (void)out_size; (void)ws_size;

  const float* x     = (const float*)d_in[0];
  const float* Wb    = (const float*)d_in[1];
  const float* bb    = (const float*)d_in[2];
  const float* Wc    = (const float*)d_in[3];
  const float* bc    = (const float*)d_in[4];
  const float* Wd    = (const float*)d_in[5];
  const float* bd    = (const float*)d_in[6];
  const float* alpha = (const float*)d_in[7];
  float* out = (float*)d_out;

  // Workspace layout: Q (2 MiB) | K (2 MiB) | V (16 MiB), all f16
  const size_t QBYTES = (size_t)K_BATCH * K_N * K_CO * sizeof(f16);
  char* ws = (char*)d_ws;
  f16* Qbuf = (f16*)(ws);
  f16* Kbuf = (f16*)(ws + QBYTES);
  f16* Vbuf = (f16*)(ws + 2 * QBYTES);

  const dim3 blk(256);
  proj_f16_kernel<false><<<dim3(K_N / 128, K_CO / 64, K_BATCH), blk, 0, stream>>>(x, Wb, bb, Qbuf, K_CO);
  proj_f16_kernel<false><<<dim3(K_N / 128, K_CO / 64, K_BATCH), blk, 0, stream>>>(x, Wc, bc, Kbuf, K_CO);
  proj_f16_kernel<true ><<<dim3(K_N / 128, K_C  / 64, K_BATCH), blk, 0, stream>>>(x, Wd, bd, Vbuf, K_C);
  pam_attn_kernel<<<dim3(K_N / 32, K_BATCH), blk, 0, stream>>>(Qbuf, Kbuf, Vbuf, x, alpha, out);
}
